// AttentionBlock_8160437862793
// MI455X (gfx1250) — matrix operand order repaired and checked by reading, not compiled, not run on hardware
//
#include <hip/hip_runtime.h>

// ---------------------------------------------------------------------------
// Types for CDNA5 (gfx1250) WMMA, wave32.
// ---------------------------------------------------------------------------
typedef __attribute__((ext_vector_type(16))) __bf16        v16bf;
typedef __attribute__((ext_vector_type(8)))  float         v8f;
typedef __attribute__((ext_vector_type(4)))  unsigned int  v4u;
typedef __attribute__((ext_vector_type(8)))  unsigned int  v8u;

#define DIM       512
#define HEADS     8
#define DHEAD     64
#define INNER     512
#define BATCH     4
#define SEQ       2048
#define ROWS      (BATCH * SEQ)      // 8192
#define QKV_COLS  (3 * INNER)        // 1536

// ---------------------------------------------------------------------------
// Helpers
// ---------------------------------------------------------------------------
__device__ __forceinline__ __bf16 to_bf16(float f) {
    unsigned u = __builtin_bit_cast(unsigned, f);
    unsigned r = u + 0x7FFFu + ((u >> 16) & 1u);   // round-to-nearest-even
    unsigned short h = (unsigned short)(r >> 16);
    return __builtin_bit_cast(__bf16, h);
}

// A-fragment: two 16B chunks (K k..k+7 / k+8..k+15: callers pass p0+8, 16 contiguous K).
__device__ __forceinline__ v16bf load_frag_split(const __bf16* p0, const __bf16* p1) {
    v4u lo = *(const v4u*)p0;
    v4u hi = *(const v4u*)p1;
    v8u w = {lo[0], lo[1], lo[2], lo[3], hi[0], hi[1], hi[2], hi[3]};
    return __builtin_bit_cast(v16bf, w);
}

// B-fragment: 32 contiguous bytes per lane (16 bf16 along K).
__device__ __forceinline__ v16bf load_frag32(const __bf16* p) {
    return __builtin_bit_cast(v16bf, *(const v8u*)p);
}

__device__ __forceinline__ v8f wmma_bf16(v16bf a, v16bf b, v8f c) {
    // (neg_a, A, neg_b, B, c_mod, C, reuse_a, reuse_b)
    return __builtin_amdgcn_wmma_f32_16x16x32_bf16(false, a, false, b,
                                                   (short)0, c, false, false);
}

// ---------------------------------------------------------------------------
// fp32 -> bf16 conversions
// ---------------------------------------------------------------------------
__global__ void cvt_bf16_kernel(const float* __restrict__ in,
                                __bf16* __restrict__ out, int n) {
    int i = blockIdx.x * blockDim.x + threadIdx.x;
    if (i < n) out[i] = to_bf16(in[i]);
}

// out[c*rows + r] = in[r*cols + c]  (store weight transposed, K contiguous)
__global__ void cvt_transpose_kernel(const float* __restrict__ in,
                                     __bf16* __restrict__ out,
                                     int rows, int cols) {
    int c = blockIdx.x * 16 + threadIdx.x;
    int r = blockIdx.y * 16 + threadIdx.y;
    if (r < rows && c < cols)
        out[(size_t)c * rows + r] = to_bf16(in[(size_t)r * cols + c]);
}

// ---------------------------------------------------------------------------
// QKV projection GEMM: [8192,512](bf16) x WT[1536,512](bf16) -> Q/K/VT (bf16)
// Block = 128 threads = 4 waves; each wave computes a 32(M) x 64(N) tile.
// The k-loop is FULLY UNROLLED with a one-iteration-ahead fragment preload:
// full unrolling lets copy-propagation kill the double-buffer rotation movs
// (and their WMMA hazard NOPs) while keeping loads a full iteration ahead of
// the WMMAs that consume them.
// ---------------------------------------------------------------------------
__global__ __launch_bounds__(128) void gemm_qkv_kernel(
    const __bf16* __restrict__ X, const __bf16* __restrict__ WT,
    __bf16* __restrict__ Qo, __bf16* __restrict__ Ko, __bf16* __restrict__ VTo)
{
    const int lane = threadIdx.x & 31;
    const int wave = threadIdx.x >> 5;
    const int hf   = lane >> 4;     // lane-half per WMMA operand layout
    const int l16  = lane & 15;
    const int m0   = blockIdx.x * 128 + wave * 32;
    const int n0   = blockIdx.y * 64;

    v8f acc[8] = {{}, {}, {}, {}, {}, {}, {}, {}};
    const __bf16* xr0 = X + (size_t)(m0 + l16) * DIM + hf * 16;
    const __bf16* xr1 = xr0 + (size_t)16 * DIM;

    // Prime the pipeline: fragments for k0 = 0.
    v16bf a0 = load_frag_split(xr0, xr0 + 8);
    v16bf a1 = load_frag_split(xr1, xr1 + 8);
    v16bf bcur[4];
#pragma unroll
    for (int j = 0; j < 4; j++)
        bcur[j] = load_frag32(WT + (size_t)(n0 + j * 16 + l16) * DIM + hf * 16);

#pragma unroll
    for (int k0 = 0; k0 < DIM; k0 += 32) {
        const bool more = (k0 + 32) < DIM;
        // Issue next k-step's loads first (consumed next iteration).
        v16bf a0n = a0, a1n = a1, bnxt[4];
#pragma unroll
        for (int j = 0; j < 4; j++) bnxt[j] = bcur[j];
        if (more) {
            a0n = load_frag_split(xr0 + k0 + 32, xr0 + k0 + 40);
            a1n = load_frag_split(xr1 + k0 + 32, xr1 + k0 + 40);
#pragma unroll
            for (int j = 0; j < 4; j++)
                bnxt[j] = load_frag32(
                    WT + (size_t)(n0 + j * 16 + l16) * DIM + k0 + 32 + hf * 16);
        }
#pragma unroll
        for (int j = 0; j < 4; j++) {
            acc[j]     = wmma_bf16(a0, bcur[j], acc[j]);
            acc[4 + j] = wmma_bf16(a1, bcur[j], acc[4 + j]);
        }
        a0 = a0n; a1 = a1n;
#pragma unroll
        for (int j = 0; j < 4; j++) bcur[j] = bnxt[j];
    }

    const int sel = n0 >> 9;        // 0=q 1=k 2=v (uniform per 64-wide tile)
#pragma unroll
    for (int j = 0; j < 4; j++) {
        int c = (n0 & 511) + j * 16 + l16;
        int h = c >> 6, d = c & 63;
#pragma unroll
        for (int half_m = 0; half_m < 2; half_m++) {
#pragma unroll
            for (int r = 0; r < 8; r++) {
                int m = m0 + half_m * 16 + hf * 8 + r;   // C layout rows
                int b = m >> 11, nn = m & 2047;
                __bf16 v = to_bf16(acc[half_m * 4 + j][r]);
                if (sel == 0)
                    Qo[(((size_t)b * HEADS + h) * SEQ + nn) * DHEAD + d] = v;
                else if (sel == 1)
                    Ko[(((size_t)b * HEADS + h) * SEQ + nn) * DHEAD + d] = v;
                else  // V transposed [b,h,d,n]: P@V B-frags become contiguous
                    VTo[(((size_t)b * HEADS + h) * DHEAD + d) * SEQ + nn] = v;
            }
        }
    }
}

// ---------------------------------------------------------------------------
// Flash attention: one (b,h) per blockIdx.y; 64 query rows per block
// (16 per wave). Online softmax, 32 keys/iter, pipelined K/V fragment loads:
//  - V frags issued at top of iteration (fly during softmax + LDS reshape)
//  - next K frags issued right after the S WMMAs (loaded directly into the
//    loop-carried registers, so no rotation copies)
// ---------------------------------------------------------------------------
__global__ __launch_bounds__(128) void flash_attn_kernel(
    const __bf16* __restrict__ Q, const __bf16* __restrict__ K,
    const __bf16* __restrict__ VT, __bf16* __restrict__ O)
{
    const int lane = threadIdx.x & 31;
    const int wave = threadIdx.x >> 5;
    const int hf   = lane >> 4;
    const int l16  = lane & 15;
    const int bh   = blockIdx.y;                 // 0..31
    const int m0   = blockIdx.x * 64 + wave * 16;

    const __bf16* Qb = Q  + (size_t)bh * SEQ * DHEAD;
    const __bf16* Kb = K  + (size_t)bh * SEQ * DHEAD;
    const __bf16* Vb = VT + (size_t)bh * DHEAD * SEQ;

    // Per-wave LDS staging for the P tile (16 rows x 32 keys, padded stride).
    __shared__ __bf16 ldsP[4][16 * 40];
    __bf16* myP = &ldsP[wave][0];

    // Preload Q A-fragments for d = 0..31 and 32..63 (held all loop).
    const __bf16* qrow = Qb + (size_t)(m0 + l16) * DHEAD;
    v16bf qf0 = load_frag_split(qrow + hf * 16,      qrow + hf * 16 + 8);
    v16bf qf1 = load_frag_split(qrow + 32 + hf * 16, qrow + 32 + hf * 16 + 8);

    v8f o0 = {}, o1 = {}, o2 = {}, o3 = {};
    float mstate[8], lstate[8];
#pragma unroll
    for (int r = 0; r < 8; r++) { mstate[r] = -1e30f; lstate[r] = 0.0f; }
    const float scale = 0.125f;                  // 1/sqrt(64)

    // Prime the K-fragment pipeline (keys 0..31).
    v16bf kf0, kf1, kf2, kf3;
    {
        const __bf16* kr0 = Kb + (size_t)l16 * DHEAD + hf * 16;
        const __bf16* kr1 = Kb + (size_t)(16 + l16) * DHEAD + hf * 16;
        kf0 = load_frag32(kr0); kf1 = load_frag32(kr0 + 32);
        kf2 = load_frag32(kr1); kf3 = load_frag32(kr1 + 32);
    }

    for (int j0 = 0; j0 < SEQ; j0 += 32) {
        // ---- V fragments: independent of softmax, issue early -------------
        const __bf16* vb = Vb + j0 + hf * 16;
        v16bf vf0 = load_frag32(vb + (size_t)(l16)      * SEQ);
        v16bf vf1 = load_frag32(vb + (size_t)(16 + l16) * SEQ);
        v16bf vf2 = load_frag32(vb + (size_t)(32 + l16) * SEQ);
        v16bf vf3 = load_frag32(vb + (size_t)(48 + l16) * SEQ);

        // ---- S tiles: keys [j0, j0+16) and [j0+16, j0+32) -----------------
        v8f s0 = {}, s1 = {};
        s0 = wmma_bf16(qf0, kf0, s0);
        s0 = wmma_bf16(qf1, kf1, s0);
        s1 = wmma_bf16(qf0, kf2, s1);
        s1 = wmma_bf16(qf1, kf3, s1);

        // ---- prefetch next key block's K fragments ------------------------
        if (j0 + 32 < SEQ) {
            const __bf16* kr0 = Kb + (size_t)(j0 + 32 + l16) * DHEAD + hf * 16;
            const __bf16* kr1 = Kb + (size_t)(j0 + 48 + l16) * DHEAD + hf * 16;
            kf0 = load_frag32(kr0); kf1 = load_frag32(kr0 + 32);
            kf2 = load_frag32(kr1); kf3 = load_frag32(kr1 + 32);
        }

        // ---- online softmax (stats replicated in C-layout row order) ------
        float mr[8];
#pragma unroll
        for (int r = 0; r < 8; r++) {
            s0[r] *= scale; s1[r] *= scale;
            mr[r] = fmaxf(s0[r], s1[r]);
        }
#pragma unroll
        for (int msk = 1; msk < 16; msk <<= 1)
#pragma unroll
            for (int r = 0; r < 8; r++)
                mr[r] = fmaxf(mr[r], __shfl_xor(mr[r], msk, 32));

        float p0[8], p1[8], corr[8], rs[8];
#pragma unroll
        for (int r = 0; r < 8; r++) {
            float mnew = fmaxf(mstate[r], mr[r]);
            corr[r]   = __expf(mstate[r] - mnew);
            mstate[r] = mnew;
            p0[r] = __expf(s0[r] - mnew);
            p1[r] = __expf(s1[r] - mnew);
            rs[r] = p0[r] + p1[r];
        }
#pragma unroll
        for (int msk = 1; msk < 16; msk <<= 1)
#pragma unroll
            for (int r = 0; r < 8; r++)
                rs[r] += __shfl_xor(rs[r], msk, 32);
#pragma unroll
        for (int r = 0; r < 8; r++) {
            lstate[r] = lstate[r] * corr[r] + rs[r];
            o0[r] *= corr[r]; o1[r] *= corr[r];
            o2[r] *= corr[r]; o3[r] *= corr[r];
        }

        // ---- reshape P (C layout) -> A fragment via LDS -------------------
#pragma unroll
        for (int r = 0; r < 8; r++) {
            int row = hf * 8 + r;
            myP[row * 40 + l16]      = to_bf16(p0[r]);
            myP[row * 40 + 16 + l16] = to_bf16(p1[r]);
        }
        __syncthreads();
        const __bf16* pr = myP + l16 * 40 + hf * 16;
        v16bf pf = load_frag_split(pr, pr + 8);
        __syncthreads();

        // ---- O += P @ V  (V frags already resident) -----------------------
        o0 = wmma_bf16(pf, vf0, o0);
        o1 = wmma_bf16(pf, vf1, o1);
        o2 = wmma_bf16(pf, vf2, o2);
        o3 = wmma_bf16(pf, vf3, o3);
    }

    // ---- finalize: O /= l, write to [8192, 512] bf16 ----------------------
    const int b = bh >> 3, h = bh & 7;
#pragma unroll
    for (int r = 0; r < 8; r++) {
        float inv = 1.0f / lstate[r];
        int n = m0 + hf * 8 + r;
        size_t base = ((size_t)b * SEQ + n) * INNER + h * DHEAD;
        O[base +      l16] = to_bf16(o0[r] * inv);
        O[base + 16 + l16] = to_bf16(o1[r] * inv);
        O[base + 32 + l16] = to_bf16(o2[r] * inv);
        O[base + 48 + l16] = to_bf16(o3[r] * inv);
    }
}

// ---------------------------------------------------------------------------
// Output projection: [8192,512](bf16) x WoutT[512,512](bf16) + bias -> f32
// Same fully-unrolled, preloaded scheme as the QKV GEMM.
// ---------------------------------------------------------------------------
__global__ __launch_bounds__(128) void gemm_out_kernel(
    const __bf16* __restrict__ A, const __bf16* __restrict__ WT,
    const float* __restrict__ bias, float* __restrict__ Out)
{
    const int lane = threadIdx.x & 31;
    const int wave = threadIdx.x >> 5;
    const int hf   = lane >> 4;
    const int l16  = lane & 15;
    const int m0   = blockIdx.x * 128 + wave * 32;
    const int n0   = blockIdx.y * 64;

    v8f acc[8] = {{}, {}, {}, {}, {}, {}, {}, {}};
    const __bf16* ar0 = A + (size_t)(m0 + l16) * INNER + hf * 16;
    const __bf16* ar1 = ar0 + (size_t)16 * INNER;

    v16bf a0 = load_frag_split(ar0, ar0 + 8);
    v16bf a1 = load_frag_split(ar1, ar1 + 8);
    v16bf bcur[4];
#pragma unroll
    for (int j = 0; j < 4; j++)
        bcur[j] = load_frag32(WT + (size_t)(n0 + j * 16 + l16) * INNER + hf * 16);

#pragma unroll
    for (int k0 = 0; k0 < INNER; k0 += 32) {
        const bool more = (k0 + 32) < INNER;
        v16bf a0n = a0, a1n = a1, bnxt[4];
#pragma unroll
        for (int j = 0; j < 4; j++) bnxt[j] = bcur[j];
        if (more) {
            a0n = load_frag_split(ar0 + k0 + 32, ar0 + k0 + 40);
            a1n = load_frag_split(ar1 + k0 + 32, ar1 + k0 + 40);
#pragma unroll
            for (int j = 0; j < 4; j++)
                bnxt[j] = load_frag32(
                    WT + (size_t)(n0 + j * 16 + l16) * INNER + k0 + 32 + hf * 16);
        }
#pragma unroll
        for (int j = 0; j < 4; j++) {
            acc[j]     = wmma_bf16(a0, bcur[j], acc[j]);
            acc[4 + j] = wmma_bf16(a1, bcur[j], acc[4 + j]);
        }
        a0 = a0n; a1 = a1n;
#pragma unroll
        for (int j = 0; j < 4; j++) bcur[j] = bnxt[j];
    }

#pragma unroll
    for (int j = 0; j < 4; j++) {
        int n = n0 + j * 16 + l16;
        float bb = bias[n];
#pragma unroll
        for (int half_m = 0; half_m < 2; half_m++) {
#pragma unroll
            for (int r = 0; r < 8; r++) {
                int m = m0 + half_m * 16 + hf * 8 + r;
                // streaming store: final output is never re-read on device
                __builtin_nontemporal_store(acc[half_m * 4 + j][r] + bb,
                                            &Out[(size_t)m * DIM + n]);
            }
        }
    }
}

// ---------------------------------------------------------------------------
// Launch
// ---------------------------------------------------------------------------
extern "C" void kernel_launch(void* const* d_in, const int* in_sizes, int n_in,
                              void* d_out, int out_size, void* d_ws, size_t ws_size,
                              hipStream_t stream) {
    (void)in_sizes; (void)n_in; (void)out_size; (void)ws_size;
    const float* x     = (const float*)d_in[0];   // [4,2048,512]
    const float* w_qkv = (const float*)d_in[1];   // [512,1536]
    const float* w_out = (const float*)d_in[2];   // [512,512]
    const float* b_out = (const float*)d_in[3];   // [512]
    float* out = (float*)d_out;                   // [4,2048,512]

    // Workspace carve-up (bf16 intermediates, ~44 MB total)
    __bf16* xb    = (__bf16*)d_ws;                       // 8192*512
    __bf16* wqkvT = xb    + (size_t)ROWS * DIM;          // 1536*512
    __bf16* qb    = wqkvT + (size_t)QKV_COLS * DIM;      // 32*2048*64
    __bf16* kb    = qb    + (size_t)BATCH * HEADS * SEQ * DHEAD;
    __bf16* vtb   = kb    + (size_t)BATCH * HEADS * SEQ * DHEAD;
    __bf16* ob    = vtb   + (size_t)BATCH * HEADS * SEQ * DHEAD;   // 8192*512
    __bf16* woutT = ob    + (size_t)ROWS * INNER;        // 512*512

    // 1) precision conversion (+ weight transposes for contiguous K loads)
    {
        int n = ROWS * DIM;
        cvt_bf16_kernel<<<(n + 255) / 256, 256, 0, stream>>>(x, xb, n);
        cvt_transpose_kernel<<<dim3(QKV_COLS / 16, DIM / 16), dim3(16, 16), 0,
                               stream>>>(w_qkv, wqkvT, DIM, QKV_COLS);
        cvt_transpose_kernel<<<dim3(DIM / 16, INNER / 16), dim3(16, 16), 0,
                               stream>>>(w_out, woutT, INNER, DIM);
    }
    // 2) QKV projection (WMMA bf16, 32x64/wave, unrolled + preloaded)
    gemm_qkv_kernel<<<dim3(ROWS / 128, QKV_COLS / 64), 128, 0, stream>>>(
        xb, wqkvT, qb, kb, vtb);
    // 3) flash attention (WMMA bf16, online softmax, pipelined K/V frags)
    flash_attn_kernel<<<dim3(SEQ / 64, BATCH * HEADS), 128, 0, stream>>>(
        qb, kb, vtb, ob);
    // 4) output projection + bias (WMMA bf16, f32 nontemporal out)
    gemm_out_kernel<<<dim3(ROWS / 128, DIM / 64), 128, 0, stream>>>(
        ob, woutT, b_out, out);
}
